// MultiHeadAttention_6854767804979
// MI455X (gfx1250) — matrix operand order repaired and checked by reading, not compiled, not run on hardware
//
#include <hip/hip_runtime.h>

typedef unsigned short u16;
typedef unsigned int   u32;
typedef __attribute__((ext_vector_type(16))) __bf16 v16bf;
typedef __attribute__((ext_vector_type(8)))  float  v8f;

#define BB   4
#define SSQ  2048
#define EE   768
#define HHD  12
#define DDD  64
#define ROWS (BB*SSQ)      /* 8192 */
#define FQKV (3*EE)        /* 2304 */
#define ATT_SCALE 0.125f   /* 64^-0.5 */

// ---------- gfx1250 async global->LDS (guarded; fallback = sync copy) ----------
#if defined(__has_builtin)
#if __has_builtin(__builtin_amdgcn_global_load_async_to_lds_b128)
#define HAVE_ASYNC_LDS 1
#endif
#if __has_builtin(__builtin_amdgcn_s_wait_asynccnt)
#define HAVE_WAIT_ASYNC 1
#endif
#endif

// builtin signature (from compiler diagnostic): param0 = v4i addrspace(1)*,
// param1 = v4i addrspace(3)*, then i32 offset, i32 cpol
typedef __attribute__((vector_size(16))) int v4i_t;
typedef __attribute__((address_space(1))) v4i_t* gv4p;
typedef __attribute__((address_space(3))) v4i_t* lv4p;

static __device__ __forceinline__ void async_cp16(const u16* g, u16* l) {
#if defined(HAVE_ASYNC_LDS)
  __builtin_amdgcn_global_load_async_to_lds_b128((gv4p)(const void*)g,
                                                 (lv4p)(void*)l, 0, 0);
#else
  __builtin_memcpy(l, g, 16);   // global_load_b128 + ds_store_b128
#endif
}
static __device__ __forceinline__ void wait_async() {
#if defined(HAVE_ASYNC_LDS)
#if defined(HAVE_WAIT_ASYNC)
  __builtin_amdgcn_s_wait_asynccnt(0);
#else
  asm volatile("s_wait_asynccnt 0" ::: "memory");
#endif
#endif
}

// ---------- helpers ----------
static __device__ __forceinline__ u16 f2bf(float f) {
  u32 u = __builtin_bit_cast(u32, f);
  u32 r = u + 0x7FFFu + ((u >> 16) & 1u);   // round-to-nearest-even
  return (u16)(r >> 16);
}

static __device__ __forceinline__ v8f zero8() {
  v8f z = {0.f,0.f,0.f,0.f,0.f,0.f,0.f,0.f};
  return z;
}

// load 16 bf16 (two 16B chunks) into a WMMA fragment register set
static __device__ __forceinline__ v16bf load_frag(const u16* lo, const u16* hi) {
  v16bf v;
  __builtin_memcpy(reinterpret_cast<char*>(&v),      lo, 16);
  __builtin_memcpy(reinterpret_cast<char*>(&v) + 16, hi, 16);
  return v;
}

// A-matrix 16x32 bf16 fragment (ISA 7.12.2): lane = {hh=lane/16, m=lane%16}
// row points at this lane's row; halves at k = 8*hh and k = 16 + 8*hh
static __device__ __forceinline__ v16bf load_A(const u16* row, int hh) {
  return load_frag(row + 8*hh, row + 16 + 8*hh);
}
// B-matrix 32x16 bf16 fragment: lane = {hh, n}; halves at k = 8*hh and k = 16 + 8*hh
static __device__ __forceinline__ v16bf load_B(const u16* row, int hh) {
  return load_frag(row + 8*hh, row + 16 + 8*hh);
}

static __device__ __forceinline__ v8f wmma_bf16(v16bf a, v16bf b, v8f c) {
  // (neg_a, A, neg_b, B, c_mod, C, reuse_a, reuse_b)
  return __builtin_amdgcn_wmma_f32_16x16x32_bf16(false, a, false, b, (short)0, c,
                                                 false, false);
}

// ---------- kernels ----------
__global__ void cvt_bf16(const float* __restrict__ in, u16* __restrict__ out, int n) {
  int i = blockIdx.x * blockDim.x + threadIdx.x;
  int stride = gridDim.x * blockDim.x;
  for (; i < n; i += stride) out[i] = f2bf(in[i]);
}

// C[ROWS x FQKV] = Xbf * Wqkv^T; 64x64 per wave (4x4 register blocking);
// scatter into q,k [B,H,S,D] and v^T [B,H,D,S] bf16
__global__ void __launch_bounds__(256) qkv_gemm(const u16* __restrict__ xb,
                                                const u16* __restrict__ wb,
                                                u16* __restrict__ qb,
                                                u16* __restrict__ kb,
                                                u16* __restrict__ vt) {
  const int lane = threadIdx.x & 31;
  const int wave = threadIdx.x >> 5;
  const int tile = blockIdx.x * (blockDim.x >> 5) + wave;
  const int ntf  = FQKV / 64;               // 36 tiles along f
  const int mt   = tile / ntf;
  const int nt   = tile % ntf;
  if (mt >= ROWS / 64) return;              // wave-uniform
  const int hh = lane >> 4, ln = lane & 15;
  const int M0 = mt * 64, N0 = nt * 64;

  const u16* arow[4];
  const u16* brow[4];
#pragma unroll
  for (int i = 0; i < 4; ++i) arow[i] = xb + (size_t)(M0 + i*16 + ln) * EE;
#pragma unroll
  for (int j = 0; j < 4; ++j) brow[j] = wb + (size_t)(N0 + j*16 + ln) * EE;

  v8f acc[4][4];
#pragma unroll
  for (int i = 0; i < 4; ++i)
#pragma unroll
    for (int j = 0; j < 4; ++j) acc[i][j] = zero8();

  for (int kk = 0; kk < EE; kk += 32) {
    v16bf a[4];
#pragma unroll
    for (int i = 0; i < 4; ++i) a[i] = load_A(arow[i] + kk, hh);
#pragma unroll
    for (int j = 0; j < 4; ++j) {
      v16bf b = load_B(brow[j] + kk, hh);
#pragma unroll
      for (int i = 0; i < 4; ++i) acc[i][j] = wmma_bf16(a[i], b, acc[i][j]);
    }
  }

  // 64-wide N tile is head-aligned: one (which, h) per wave
  const int which = N0 / EE;
  const int h     = (N0 % EE) / DDD;
#pragma unroll
  for (int j = 0; j < 4; ++j) {
    const int d = j*16 + ln;                // d within head
#pragma unroll
    for (int i = 0; i < 4; ++i) {
#pragma unroll
      for (int jj = 0; jj < 8; ++jj) {
        int row = M0 + i*16 + jj + 8*hh;    // global sequence-row
        int b_  = row / SSQ, s = row % SSQ;
        u16 val = f2bf(acc[i][j][jj]);
        if (which == 0)      qb[(size_t)(((b_*HHD + h)*SSQ) + s)*DDD + d] = val;
        else if (which == 1) kb[(size_t)(((b_*HHD + h)*SSQ) + s)*DDD + d] = val;
        else                 vt[(size_t)(((b_*HHD + h)*DDD) + d)*SSQ + s] = val;
      }
    }
  }
}

// flash attention: block = 8 waves on the same (b,h), 256 queries.
// K/V blocks staged once per workgroup into LDS (async DMA, double-buffered);
// each wave computes 2 query tiles from the shared LDS copy.
__global__ void __launch_bounds__(256) flash_attn(const u16* __restrict__ qb,
                                                  const u16* __restrict__ kb,
                                                  const u16* __restrict__ vt,
                                                  u16* __restrict__ ob) {
  __shared__ u16 kbuf[2][32*DDD];           // [key][d]   4KB per buffer
  __shared__ u16 vbuf[2][DDD*32];           // [d][key]   4KB per buffer
  __shared__ u16 plds[8][2][16*32];         // per-wave P bounce (16KB)

  const int tid  = threadIdx.x;
  const int lane = tid & 31;
  const int wave = tid >> 5;
  const int gw   = blockIdx.x * 8 + wave;
  const int qblk = gw % (SSQ/32);           // 64 query-blocks of 32 (8 | 64 ->
  const int bh   = gw / (SSQ/32);           //  all waves in block share bh)
  const int hh = lane >> 4, ln = lane & 15;

  const u16* Kb = kb + (size_t)bh * SSQ * DDD;
  const u16* Vb = vt + (size_t)bh * DDD * SSQ;

  v16bf qa[2][2];
#pragma unroll
  for (int t = 0; t < 2; ++t) {
    const u16* Qrow = qb + (size_t)(bh*SSQ + qblk*32 + t*16 + ln) * DDD;
    qa[t][0] = load_A(Qrow +  0, hh);       // d 0..31
    qa[t][1] = load_A(Qrow + 32, hh);       // d 32..63
  }

  float mrun[2][8], lrun[2][8], alpha[2][8];
  v8f acc[2][4];
#pragma unroll
  for (int t = 0; t < 2; ++t) {
#pragma unroll
    for (int j = 0; j < 8; ++j) { mrun[t][j] = -1e30f; lrun[t][j] = 0.f; }
#pragma unroll
    for (int c = 0; c < 4; ++c) acc[t][c] = zero8();
  }

  // cooperative stage of one 32-key block: 256 threads x 16B for K, same for V
  auto stage = [&](int kbk, int buf) {
    async_cp16(Kb + (size_t)kbk*DDD + tid*8, &kbuf[buf][tid*8]);       // contiguous 4KB
    int vr = tid >> 2, vch = tid & 3;                                  // 64 rows x 64B
    async_cp16(Vb + (size_t)vr*SSQ + kbk + vch*8, &vbuf[buf][vr*32 + vch*8]);
  };

  stage(0, 0);
  wait_async();
  __syncthreads();

  for (int it = 0; it < SSQ/32; ++it) {
    const int kbk = it * 32;
    const int cur = it & 1;
    if (it + 1 < SSQ/32) stage(kbk + 32, cur ^ 1);   // fill next while computing

    const u16* kc = &kbuf[cur][0];
    const u16* vv = &vbuf[cur][0];

    // ---- K fragments from LDS (shared by all waves & both q-tiles) ----
    v16bf kf00 = load_B(kc + (      ln)*DDD +  0, hh);
    v16bf kf01 = load_B(kc + (      ln)*DDD + 32, hh);
    v16bf kf10 = load_B(kc + (16 + ln)*DDD +  0, hh);
    v16bf kf11 = load_B(kc + (16 + ln)*DDD + 32, hh);

#pragma unroll
    for (int t = 0; t < 2; ++t) {
      // ---- S = Q * K^T ----
      v8f s0 = zero8(), s1 = zero8();
      s0 = wmma_bf16(qa[t][0], kf00, s0);
      s0 = wmma_bf16(qa[t][1], kf01, s0);
      s1 = wmma_bf16(qa[t][0], kf10, s1);
      s1 = wmma_bf16(qa[t][1], kf11, s1);

      // ---- online softmax (rows = j + 8*hh; cols across 16-lane half) ----
      u16* P = &plds[wave][t][0];
#pragma unroll
      for (int j = 0; j < 8; ++j) {
        float a0 = s0[j] * ATT_SCALE;
        float a1 = s1[j] * ATT_SCALE;
        float mx = fmaxf(a0, a1);
        mx = fmaxf(mx, __shfl_xor(mx, 1, 32));
        mx = fmaxf(mx, __shfl_xor(mx, 2, 32));
        mx = fmaxf(mx, __shfl_xor(mx, 4, 32));
        mx = fmaxf(mx, __shfl_xor(mx, 8, 32));
        float mnew = fmaxf(mrun[t][j], mx);
        float al   = __expf(mrun[t][j] - mnew);
        float p0   = __expf(a0 - mnew);
        float p1   = __expf(a1 - mnew);
        float rs = p0 + p1;
        rs += __shfl_xor(rs, 1, 32);
        rs += __shfl_xor(rs, 2, 32);
        rs += __shfl_xor(rs, 4, 32);
        rs += __shfl_xor(rs, 8, 32);
        lrun[t][j] = lrun[t][j] * al + rs;
        mrun[t][j] = mnew;
        alpha[t][j] = al;
        P[(j + 8*hh)*32 + ln]      = f2bf(p0);
        P[(j + 8*hh)*32 + 16 + ln] = f2bf(p1);
      }
#pragma unroll
      for (int c = 0; c < 4; ++c)
#pragma unroll
        for (int j = 0; j < 8; ++j) acc[t][c][j] *= alpha[t][j];
    }

    __builtin_amdgcn_wave_barrier();
    v16bf pa[2];
    pa[0] = load_A(&plds[wave][0][0] + ln*32, hh);
    pa[1] = load_A(&plds[wave][1][0] + ln*32, hh);
    __builtin_amdgcn_wave_barrier();

    // ---- O += P * V from LDS ([d][key] rows, 32-key stride) ----
#pragma unroll
    for (int c = 0; c < 4; ++c) {
      v16bf vf = load_B(vv + (c*16 + ln)*32, hh);
      acc[0][c] = wmma_bf16(pa[0], vf, acc[0][c]);
      acc[1][c] = wmma_bf16(pa[1], vf, acc[1][c]);
    }

    wait_async();        // next block's DMA landed
    __syncthreads();     // everyone done reading cur before it's refilled
  }

  // ---- normalize + write bf16 attn-out in [B,S,E] ----
  const int b_ = bh / HHD, h = bh % HHD;
#pragma unroll
  for (int t = 0; t < 2; ++t)
#pragma unroll
    for (int c = 0; c < 4; ++c)
#pragma unroll
      for (int j = 0; j < 8; ++j) {
        float o = acc[t][c][j] / lrun[t][j];
        int s = qblk*32 + t*16 + j + 8*hh;
        ob[(size_t)(b_*SSQ + s)*EE + h*DDD + c*16 + ln] = f2bf(o);
      }
}

// out[ROWS x EE] = Abf * Wproj^T + bias (fp32 out); 64x64 per wave
__global__ void __launch_bounds__(256) proj_gemm(const u16* __restrict__ ab,
                                                 const u16* __restrict__ wb,
                                                 const float* __restrict__ bias,
                                                 float* __restrict__ out) {
  const int lane = threadIdx.x & 31;
  const int wave = threadIdx.x >> 5;
  const int tile = blockIdx.x * (blockDim.x >> 5) + wave;
  const int ntf  = EE / 64;                 // 12
  const int mt   = tile / ntf;
  const int nt   = tile % ntf;
  if (mt >= ROWS / 64) return;
  const int hh = lane >> 4, ln = lane & 15;
  const int M0 = mt * 64, N0 = nt * 64;

  const u16* arow[4];
  const u16* brow[4];
#pragma unroll
  for (int i = 0; i < 4; ++i) arow[i] = ab + (size_t)(M0 + i*16 + ln) * EE;
#pragma unroll
  for (int j = 0; j < 4; ++j) brow[j] = wb + (size_t)(N0 + j*16 + ln) * EE;

  v8f acc[4][4];
#pragma unroll
  for (int i = 0; i < 4; ++i)
#pragma unroll
    for (int j = 0; j < 4; ++j) acc[i][j] = zero8();

  for (int kk = 0; kk < EE; kk += 32) {
    v16bf a[4];
#pragma unroll
    for (int i = 0; i < 4; ++i) a[i] = load_A(arow[i] + kk, hh);
#pragma unroll
    for (int j = 0; j < 4; ++j) {
      v16bf b = load_B(brow[j] + kk, hh);
#pragma unroll
      for (int i = 0; i < 4; ++i) acc[i][j] = wmma_bf16(a[i], b, acc[i][j]);
    }
  }

#pragma unroll
  for (int j = 0; j < 4; ++j) {
    const float bv = bias[N0 + j*16 + ln];
#pragma unroll
    for (int i = 0; i < 4; ++i) {
#pragma unroll
      for (int jj = 0; jj < 8; ++jj) {
        int row = M0 + i*16 + jj + 8*hh;
        out[(size_t)row*EE + N0 + j*16 + ln] = acc[i][j][jj] + bv;
      }
    }
  }
}

// ---------- host ----------
extern "C" void kernel_launch(void* const* d_in, const int* in_sizes, int n_in,
                              void* d_out, int out_size, void* d_ws, size_t ws_size,
                              hipStream_t stream) {
  (void)in_sizes; (void)n_in; (void)out_size; (void)ws_size;
  const float* x      = (const float*)d_in[0];
  const float* w_qkv  = (const float*)d_in[1];
  const float* w_proj = (const float*)d_in[2];
  const float* b_proj = (const float*)d_in[3];
  float* out = (float*)d_out;

  u16* p = (u16*)d_ws;
  u16* xb     = p; p += (size_t)ROWS * EE;
  u16* wqkvb  = p; p += (size_t)FQKV * EE;
  u16* wprojb = p; p += (size_t)EE * EE;
  u16* qb     = p; p += (size_t)ROWS * EE;   // [B,H,S,D]
  u16* kbw    = p; p += (size_t)ROWS * EE;   // [B,H,S,D]
  u16* vt     = p; p += (size_t)ROWS * EE;   // [B,H,D,S]
  u16* ob     = p; p += (size_t)ROWS * EE;   // [B,S,E]

  cvt_bf16<<<2048, 256, 0, stream>>>(x,      xb,     ROWS*EE);
  cvt_bf16<<<2048, 256, 0, stream>>>(w_qkv,  wqkvb,  FQKV*EE);
  cvt_bf16<<<1024, 256, 0, stream>>>(w_proj, wprojb, EE*EE);

  // 128 x 36 tiles of 64x64, 8 waves/block
  qkv_gemm<<<(ROWS/64)*(FQKV/64)/8, 256, 0, stream>>>(xb, wqkvb, qb, kbw, vt);
  // 48 bh * 64 query-blocks = 3072 waves (8 waves/block share one bh)
  flash_attn<<<(BB*HHD)*(SSQ/32)/8, 256, 0, stream>>>(qb, kbw, vt, ob);
  // 128 x 12 tiles of 64x64
  proj_gemm<<<(ROWS/64)*(EE/64)/8, 256, 0, stream>>>(ob, wprojb, b_proj, out);
}
